// sLSTMLayer_72911364817338
// MI455X (gfx1250) — matrix operand order repaired and checked by reading, not compiled, not run on hardware
//
#include <hip/hip_runtime.h>
#include <hip/hip_bf16.h>

// sLSTM (xLSTM) layer for MI455X / gfx1250, wave32.
// B=8 S=2048 D=1024 NH=4 DH=256.
// Phase 1: bf16 WMMA gate GEMM  gx[t][b][h][o*4+g] = x@W^T + bias   (~300MB, mem-bound)
// Phase 2: per-head sequential scan, recurrent h@R via v_wmma_f32_16x16x32_bf16,
//          f32 state in registers, inline group-norm, out written directly.
// Workspace use: ~306 MB.

typedef __bf16 v16bf __attribute__((ext_vector_type(16)));
typedef float  v8f   __attribute__((ext_vector_type(8)));

#define BB   8
#define SS   2048
#define DD   1024
#define NH_  4
#define DH_  256

// ---------------- x -> bf16 copy (same layout) ----------------
__global__ void k_cvt_x(const float* __restrict__ x, __bf16* __restrict__ xbf) {
  int i = (blockIdx.x * 256 + threadIdx.x) * 8;
  float4 a = *(const float4*)(x + i);
  float4 b = *(const float4*)(x + i + 4);
  union { __bf16 h[8]; uint4 u; } p;
  p.h[0] = (__bf16)a.x; p.h[1] = (__bf16)a.y; p.h[2] = (__bf16)a.z; p.h[3] = (__bf16)a.w;
  p.h[4] = (__bf16)b.x; p.h[5] = (__bf16)b.y; p.h[6] = (__bf16)b.z; p.h[7] = (__bf16)b.w;
  *(uint4*)(xbf + i) = p.u;
}

// ---------------- pack gate weights into per-lane B-fragment order ----------------
// Layout: Wpk[((h*64+nt)*8+ks)*512 + lane*16 + jj]
//   col = nt*16 + lane%16 ; g = col&3 ; o = col>>2
//   K   = ks*32 + (lane>>4)*8 + (jj&7) + (jj>>3)*16   (B 32x16 bf16: lanes 0-15 K=0..7,16..23; 16-31 K=8..15,24..31)
__global__ void k_pack_w(const float* __restrict__ Wf, const float* __restrict__ Wi,
                         const float* __restrict__ Wz, const float* __restrict__ Wo,
                         __bf16* __restrict__ Wpk) {
  int idx = blockIdx.x * 256 + threadIdx.x;          // 1,048,576 total
  int jj   = idx & 15;
  int lane = (idx >> 4) & 31;
  int ks   = (idx >> 9) & 7;
  int nt   = (idx >> 12) & 63;
  int h    = idx >> 18;
  int col  = nt * 16 + (lane & 15);
  int g = col & 3, o = col >> 2;
  int K = ks * 32 + ((lane >> 4) << 3) + (jj & 7) + ((jj >> 3) << 4);
  const float* W = (g == 0) ? Wf : (g == 1) ? Wi : (g == 2) ? Wz : Wo;
  Wpk[idx] = (__bf16)W[(h * 256 + o) * 256 + K];
}

// ---------------- pack R into per-lane B-fragment order ----------------
// Layout: Rpk[(((h*16+wv)*8+ks)*4+nc)*512 + lane*16 + jj]
//   col = wv*64 + nc*16 + lane%16 ; g = col&3 ; o = col>>2 ; K = ks*32 + (lane>>4)*8 + (jj&7) + (jj>>3)*16
//   source R[h][g][o][d=K]  (raw = h @ R : sum_d h[b,d]*R[h,g,o,d])
__global__ void k_pack_r(const float* __restrict__ R, __bf16* __restrict__ Rpk) {
  int idx = blockIdx.x * 256 + threadIdx.x;          // 1,048,576 total
  int jj   = idx & 15;
  int lane = (idx >> 4) & 31;
  int nc   = (idx >> 9) & 3;
  int ks   = (idx >> 11) & 7;
  int wv   = (idx >> 14) & 15;
  int h    = idx >> 18;
  int col  = wv * 64 + nc * 16 + (lane & 15);
  int g = col & 3, o = col >> 2;
  int K = ks * 32 + ((lane >> 4) << 3) + (jj & 7) + ((jj >> 3) << 4);
  Rpk[idx] = (__bf16)R[(((h * 4 + g) * 256 + o) << 8) + K];
}

// ---------------- gate projection GEMM (bf16 WMMA, no-LDS direct fragments) ------
// grid (128, 64, 4), block 256 (8 waves). Wave w computes 16x16 tile:
//   rows mbase..mbase+15 of (b*S+s), cols nt*16..+15 of packed (o*4+g), head z.
// Output: gx[((t*8+b)*4+h)*1024 + col]  (float, bias folded in).
__global__ __launch_bounds__(256) void k_gate_gemm(const __bf16* __restrict__ xbf,
                                                   const __bf16* __restrict__ Wpk,
                                                   const float* __restrict__ bias,
                                                   float* __restrict__ gx) {
  const int tid = threadIdx.x, wv = tid >> 5, lane = tid & 31;
  const int h = blockIdx.z, nt = blockIdx.y;
  const int mbase = (blockIdx.x * 8 + wv) * 16;
  const int col = nt * 16 + (lane & 15);
  const int g = col & 3, o = col >> 2;
  const float bv = bias[(h * 4 + g) * 256 + o];
  v8f acc = {bv, bv, bv, bv, bv, bv, bv, bv};

  // A fragment (16x32 bf16): lane row = mbase + lane%16 ; K chunks depend on lane/16
  const int mrow = mbase + (lane & 15);
  const __bf16* arow  = xbf + mrow * 1024 + h * 256 + ((lane >> 4) << 3);
  const __bf16* wbase = Wpk + (h * 64 + nt) * 8 * 512 + lane * 16;
#pragma unroll
  for (int ks = 0; ks < 8; ++ks) {
    v16bf a, bm;
    *(uint4*)&a       = *(const uint4*)(arow + ks * 32);        // K = base+0..7
    *((uint4*)&a + 1) = *(const uint4*)(arow + ks * 32 + 16);   // K = base+16..23
    bm = *(const v16bf*)(wbase + ks * 512);                     // 32B contiguous
    acc = __builtin_amdgcn_wmma_f32_16x16x32_bf16(false, a, false, bm,
                                                  (short)0, acc, false, false);
  }
  // C/D layout: VGPR j, lane l: row = j + 8*(l>=16), col = l%16
  const int r0 = (lane >> 4) << 3;
#pragma unroll
  for (int j = 0; j < 8; ++j) {
    int m = mbase + j + r0;
    int b = m >> 11, t = m & 2047;
    gx[(((t * 8 + b) * 4 + h) << 10) + col] = acc[j];
  }
}

// ---------------- sequential sLSTM scan + inline group-norm ----------------------
// grid 4 (one WG per head), block 512 (16 waves). Wave wv owns packed cols [64wv,64wv+64).
// h (bf16) lives in LDS padded to 16 rows; states c,n,m f32 in registers of update lanes.
__global__ __launch_bounds__(512) void k_scan(const float* __restrict__ gx,
                                              const __bf16* __restrict__ Rpk,
                                              const float* __restrict__ gn,
                                              float* __restrict__ out) {
  __shared__ __align__(16) __bf16 h_lds[16][256];   // 8 KB  (rows 8..15 stay zero)
  __shared__ __align__(16) float  raw[8][1024];     // 32 KB (h @ R for this step)
  __shared__ float psum[8][8], psq[8][8], stats[8][2];

  const int tid = threadIdx.x, wv = tid >> 5, lane = tid & 31;
  const int h = blockIdx.x;

  { // zero h (h0 = 0), including padding rows
    __bf16* p = &h_lds[0][0];
    for (int i = tid; i < 4096; i += 512) p[i] = (__bf16)0.f;
  }

  // update-phase mapping: thread handles channel o for batches b = (tid>>8)+2k
  const int o = tid & 255, bb = tid >> 8;
  const float wgn = 1.f + gn[h * 256 + o];
  float c[4]  = {0.f, 0.f, 0.f, 0.f};
  float nn[4] = {0.f, 0.f, 0.f, 0.f};
  float mm[4] = {0.f, 0.f, 0.f, 0.f};

  const float4* gx4   = (const float4*)gx;
  const __bf16* rbase = Rpk + (h * 16 + wv) * 32 * 512 + lane * 16;
  const v8f vz = {0.f, 0.f, 0.f, 0.f, 0.f, 0.f, 0.f, 0.f};

  __syncthreads();

  for (int t = 0; t < SS; ++t) {
    // prefetch this step's input-gate activations (independent of h -> overlaps WMMA)
    float4 g4[4];
#pragma unroll
    for (int k = 0; k < 4; ++k) {
      int b = bb + 2 * k;
      g4[k] = gx4[((t * 8 + b) * 4 + h) * 256 + o];
    }

    // ---- recurrent matmul: raw[b][col] = sum_d h[b][d] * Rpk (streamed from L2) ----
    v8f acc[4] = {vz, vz, vz, vz};
#pragma unroll
    for (int ks = 0; ks < 8; ++ks) {
      v16bf a;
      const int m  = lane & 15;
      const int kA = ks * 32 + ((lane >> 4) << 3);
      *(uint4*)&a       = *(const uint4*)&h_lds[m][kA];        // K = kA..kA+7
      *((uint4*)&a + 1) = *(const uint4*)&h_lds[m][kA + 16];   // K = kA+16..kA+23
#pragma unroll
      for (int nc = 0; nc < 4; ++nc) {
        v16bf bm = *(const v16bf*)(rbase + (ks * 4 + nc) * 512);
        acc[nc] = __builtin_amdgcn_wmma_f32_16x16x32_bf16(false, a, false, bm,
                                                          (short)0, acc[nc], false, false);
      }
    }
    if (lane < 16) {                 // rows 0..7 are real batches (lanes>=16 = padding rows)
      const int cb = wv * 64 + lane;
#pragma unroll
      for (int nc = 0; nc < 4; ++nc)
#pragma unroll
        for (int j = 0; j < 8; ++j)
          raw[j][cb + nc * 16] = acc[nc][j];
    }
    __syncthreads();

    // ---- elementwise sLSTM state update ----
    float hn[4];
#pragma unroll
    for (int k = 0; k < 4; ++k) {
      int b = bb + 2 * k;
      float4 r4 = ((const float4*)&raw[b][0])[o];      // f,i,z,o recurrent terms
      float fr  = g4[k].x + r4.x;
      float ir  = g4[k].y + r4.y;
      float zr  = g4[k].z + r4.z;
      float orw = g4[k].w + r4.w;
      float mnew = fmaxf(fr + mm[k], ir);
      float ig = __expf(ir - mnew);
      float fg = __expf(fr + mm[k] - mnew);
      float cnew = fg * c[k] + ig * tanhf(zr);
      float nnew = fg * nn[k] + ig;
      float hh = (1.f / (1.f + __expf(-orw))) * cnew / nnew;
      c[k] = cnew; nn[k] = nnew; mm[k] = mnew; hn[k] = hh;
      h_lds[b][o] = (__bf16)hh;                        // feed next step's A matrix
    }

    // ---- group-norm stats: per (b): reduce 256 channels = 8 waves x 32 lanes ----
#pragma unroll
    for (int k = 0; k < 4; ++k) {
      float s = hn[k], q = hn[k] * hn[k];
#pragma unroll
      for (int off = 16; off; off >>= 1) {
        s += __shfl_xor(s, off);
        q += __shfl_xor(q, off);
      }
      if (lane == 0) { psum[bb + 2 * k][wv & 7] = s; psq[bb + 2 * k][wv & 7] = q; }
    }
    __syncthreads();
    if (tid < 8) {
      float S = 0.f, Q = 0.f;
#pragma unroll
      for (int w = 0; w < 8; ++w) { S += psum[tid][w]; Q += psq[tid][w]; }
      float mean = S * (1.f / 256.f);
      float var  = Q * (1.f / 256.f) - mean * mean;
      stats[tid][0] = mean;
      stats[tid][1] = rsqrtf(var + 1e-5f);
    }
    __syncthreads();

    // ---- normalized output: out[b][t][h*256+o] ----
#pragma unroll
    for (int k = 0; k < 4; ++k) {
      int b = bb + 2 * k;
      out[((b * 2048 + t) << 10) + (h << 8) + o] = (hn[k] - stats[b][0]) * stats[b][1] * wgn;
    }
    // next iteration's raw/psum/stats writes are separated by the two barriers above
  }
}

extern "C" void kernel_launch(void* const* d_in, const int* in_sizes, int n_in,
                              void* d_out, int out_size, void* d_ws, size_t ws_size,
                              hipStream_t stream) {
  (void)in_sizes; (void)n_in; (void)out_size; (void)ws_size;
  const float* x    = (const float*)d_in[0];
  const float* Wf   = (const float*)d_in[1];
  const float* Wi   = (const float*)d_in[2];
  const float* Wz   = (const float*)d_in[3];
  const float* Wo   = (const float*)d_in[4];
  const float* R    = (const float*)d_in[5];
  const float* bias = (const float*)d_in[6];
  const float* gn   = (const float*)d_in[7];
  float* out = (float*)d_out;

  char* ws = (char*)d_ws;
  __bf16* xbf = (__bf16*)ws;                     //  33,554,432 B  (B*S*D bf16)
  __bf16* Wpk = (__bf16*)(ws + 33554432);        //   2,097,152 B
  __bf16* Rpk = (__bf16*)(ws + 35651584);        //   2,097,152 B
  float*  gx  = (float*)(ws + 37748736);         // 268,435,456 B  (S*B*NH*1024 f32)

  k_cvt_x    <<<dim3(16777216 / (256 * 8)), 256, 0, stream>>>(x, xbf);
  k_pack_w   <<<dim3(1048576 / 256),        256, 0, stream>>>(Wf, Wi, Wz, Wo, Wpk);
  k_pack_r   <<<dim3(1048576 / 256),        256, 0, stream>>>(R, Rpk);
  k_gate_gemm<<<dim3(128, 64, 4),           256, 0, stream>>>(xbf, Wpk, bias, gx);
  k_scan     <<<dim3(4),                    512, 0, stream>>>(gx, Rpk, gn, out);
}
